// RefinementModule_7593502179726
// MI455X (gfx1250) — matrix operand order repaired and checked by reading, not compiled, not run on hardware
//
#include <hip/hip_runtime.h>
#include <hip/hip_bf16.h>

#define NPTS 32768
#define KNN  16

typedef __attribute__((ext_vector_type(16))) _Float16 v16h;
typedef __attribute__((ext_vector_type(8)))  _Float16 v8h;
typedef __attribute__((ext_vector_type(8)))  float    v8f;

__device__ __forceinline__ v16h cat8(v8h a, v8h b) {
    return __builtin_shufflevector(a, b, 0,1,2,3,4,5,6,7,8,9,10,11,12,13,14,15);
}

// D = A*B + C, f16 inputs, f32 accumulate (v_wmma_f32_16x16x32_f16)
__device__ __forceinline__ v8f wmma32(v16h a, v16h b, v8f c) {
    return __builtin_amdgcn_wmma_f32_16x16x32_f16(false, a, false, b, (short)0, c, false, false);
}

// A-fragment (16xK tile, row-major in LDS, row = M): lane l -> row l%16,
// K halves [kb..kb+7] and [16+kb..16+kb+7], kb = 8*(l/16). Two ds_load_b128.
__device__ __forceinline__ v16h lds_a_frag(const _Float16* tile, int ld, int koff) {
    int lane = threadIdx.x & 31;
    const _Float16* p = tile + (lane & 15) * ld + koff + ((lane >> 4) << 3);
    v8h lo = *(const v8h*)p;
    v8h hi = *(const v8h*)(p + 16);
    return cat8(lo, hi);
}

// B-fragment from an N-major (transposed) tile: lane l -> row N=l%16,
// K halves [kb..kb+7] and [16+kb..16+kb+7], kb = 8*(l/16).
__device__ __forceinline__ v16h lds_b_frag(const _Float16* tile, int ld, int koff) {
    int lane = threadIdx.x & 31;
    const _Float16* p = tile + (lane & 15) * ld + koff + ((lane >> 4) << 3);
    v8h lo = *(const v8h*)p;
    v8h hi = *(const v8h*)(p + 16);
    return cat8(lo, hi);
}

// 8 contiguous edge-feature values for C_in=64: cols [c..c+7] of [xi | xj-xi]
__device__ __forceinline__ v8h ef_run64(const float* __restrict__ xip,
                                        const float* __restrict__ xjp, int c) {
    v8h r;
    if (c < 64) {
#pragma unroll
        for (int j = 0; j < 8; ++j) r[j] = (_Float16)xip[c + j];
    } else {
        int cc = c - 64;
#pragma unroll
        for (int j = 0; j < 8; ++j) r[j] = (_Float16)(xjp[cc + j] - xip[cc + j]);
    }
    return r;
}

__device__ __forceinline__ _Float16 ef_elem3(const float* xi, const float* xj, int cc) {
    float v = 0.f;
    if (cc < 3)      v = xi[cc];
    else if (cc < 6) v = xj[cc - 3] - xi[cc - 3];
    return (_Float16)v;
}

// ---------------- EdgeConv: one wave per node (16 edges = one M-tile) -------
template <int CIN>
__global__ __launch_bounds__(128) void edgeconv_kernel(
    const float* __restrict__ X,    // NPTS x CIN
    const int*   __restrict__ src,  // NPTS*KNN neighbor indices
    const float* __restrict__ wa,   // (2*CIN) x 64 row-major (KxN)
    const float* __restrict__ ba,
    const float* __restrict__ wb,   // 64 x 64
    const float* __restrict__ bb,
    float* __restrict__ Y)          // NPTS x 64
{
    constexpr int KA = (CIN == 3) ? 32 : 128;   // padded K of first GEMM
    constexpr int KSTEPS_A = KA / 32;

    __shared__ __align__(16) _Float16 waT[64][KA];   // N-major wa (zero-padded)
    __shared__ __align__(16) _Float16 wbT[64][64];   // N-major wb
    __shared__ __align__(16) _Float16 hT[4][16][64]; // per-wave relu(h1) tile
    __shared__ float sba[64], sbb[64];

    const int t = threadIdx.x;
    for (int i = t; i < 64 * KA; i += 128) {
        int n = i / KA, k = i % KA;
        float v = (k < 2 * CIN) ? wa[k * 64 + n] : 0.f;
        waT[n][k] = (_Float16)v;
    }
    for (int i = t; i < 64 * 64; i += 128) {
        int n = i / 64, k = i % 64;
        wbT[n][k] = (_Float16)wb[k * 64 + n];
    }
    if (t < 64) { sba[t] = ba[t]; sbb[t] = bb[t]; }
    __syncthreads();

    const int wave = t >> 5, lane = t & 31;
    const int row  = lane & 15;            // edge (M) row for A, N row for B/D
    const int kb   = (lane >> 4) << 3;
    const int node = blockIdx.x * 4 + wave;
    const int nbr  = src[node * KNN + row];

    // -------- gather edge features directly into A fragments (registers) ----
    v16h afrag[KSTEPS_A];
    if (CIN == 3) {
        float xi[3], xj[3];
        const float* xp = X + (size_t)node * 3;
        const float* jp = X + (size_t)nbr * 3;
#pragma unroll
        for (int j = 0; j < 3; ++j) { xi[j] = xp[j]; xj[j] = jp[j]; }
        v16h f;
#pragma unroll
        for (int j = 0; j < 8; ++j) {
            f[j]     = ef_elem3(xi, xj, kb + j);
            f[8 + j] = ef_elem3(xi, xj, 16 + kb + j);
        }
        afrag[0] = f;
    } else {
        const float* xip = X + (size_t)node * 64;
        const float* xjp = X + (size_t)nbr  * 64;
#pragma unroll
        for (int k = 0; k < KSTEPS_A; ++k) {
            int c0 = 32 * k + kb;
            afrag[k] = cat8(ef_run64(xip, xjp, c0), ef_run64(xip, xjp, c0 + 16));
        }
    }

    // -------- GEMM1: h = relu(ef @ wa + ba), store f16 tile to LDS ----------
    const int mbase = (lane >> 4) << 3;
#pragma unroll
    for (int nt = 0; nt < 4; ++nt) {
        // preload all B-fragments of this N-tile, then issue WMMAs back-to-back
        v16h bfr[KSTEPS_A];
#pragma unroll
        for (int ks = 0; ks < KSTEPS_A; ++ks)
            bfr[ks] = lds_b_frag(&waT[nt * 16][0], KA, ks * 32);
        float bv = sba[nt * 16 + row];
        v8f acc = {bv, bv, bv, bv, bv, bv, bv, bv};
#pragma unroll
        for (int ks = 0; ks < KSTEPS_A; ++ks)
            acc = wmma32(afrag[ks], bfr[ks], acc);
#pragma unroll
        for (int r = 0; r < 8; ++r)
            hT[wave][mbase + r][nt * 16 + row] = (_Float16)fmaxf(acc[r], 0.f);
    }
    asm volatile("s_wait_dscnt 0" ::: "memory");  // wave-local LDS RAW fence

    // -------- GEMM2: h2 = h @ wb + bb, then segment-max over the 16 edges ---
    // A fragments are N-tile invariant: load the whole 16x64 h-tile once.
    v16h ha[2];
#pragma unroll
    for (int ks = 0; ks < 2; ++ks)
        ha[ks] = lds_a_frag(&hT[wave][0][0], 64, ks * 32);
#pragma unroll
    for (int nt = 0; nt < 4; ++nt) {
        v16h bfr[2];
#pragma unroll
        for (int ks = 0; ks < 2; ++ks)
            bfr[ks] = lds_b_frag(&wbT[nt * 16][0], 64, ks * 32);
        float bv = sbb[nt * 16 + row];
        v8f acc = {bv, bv, bv, bv, bv, bv, bv, bv};
#pragma unroll
        for (int ks = 0; ks < 2; ++ks)
            acc = wmma32(ha[ks], bfr[ks], acc);
        float m = acc[0];
#pragma unroll
        for (int r = 1; r < 8; ++r) m = fmaxf(m, acc[r]);
        m = fmaxf(m, __shfl_xor(m, 16));          // combine M 0-7 with 8-15
        if (lane < 16) Y[(size_t)node * 64 + nt * 16 + lane] = m;
    }
}

// ---------------- MLP head: residual = relu(feats@mw1+mb1)@mw2+mb2 ----------
__global__ __launch_bounds__(128) void mlp_kernel(
    const float* __restrict__ f1, const float* __restrict__ f2,
    const float* __restrict__ f3, const float* __restrict__ points,
    const float* __restrict__ mw1, const float* __restrict__ mb1,
    const float* __restrict__ mw2, const float* __restrict__ mb2,
    float* __restrict__ pwr)
{
    __shared__ __align__(16) _Float16 w1T[16][192];   // one N-tile of mw1, N-major
    __shared__ __align__(16) _Float16 hF[64][256];    // relu(h) for 64 nodes
    __shared__ float smb1[256];
    __shared__ float smw2[256 * 3];

    const int t = threadIdx.x, wave = t >> 5, lane = t & 31;
    const int node0 = blockIdx.x * 64;
    for (int i = t; i < 256; i += 128) smb1[i] = mb1[i];
    for (int i = t; i < 256 * 3; i += 128) smw2[i] = mw2[i];

    // A fragments: 16 nodes x 192 feats = 6 K-steps, gathered from f1|f2|f3
    const int row = lane & 15, kb = (lane >> 4) << 3;
    const int nd = node0 + wave * 16 + row;
    const float* r1 = f1 + (size_t)nd * 64;
    const float* r2 = f2 + (size_t)nd * 64;
    const float* r3 = f3 + (size_t)nd * 64;
    v16h afrag[6];
#pragma unroll
    for (int k = 0; k < 6; ++k) {
        int c0 = 32 * k + kb, c1 = c0 + 16;
        const float* b0 = (c0 < 64) ? r1 : (c0 < 128) ? r2 : r3;
        const float* b1 = (c1 < 64) ? r1 : (c1 < 128) ? r2 : r3;
        v8h lo, hi;
#pragma unroll
        for (int j = 0; j < 8; ++j) lo[j] = (_Float16)b0[(c0 & 63) + j];
#pragma unroll
        for (int j = 0; j < 8; ++j) hi[j] = (_Float16)b1[(c1 & 63) + j];
        afrag[k] = cat8(lo, hi);
    }
    __syncthreads();

    const int mbase = (lane >> 4) << 3;
    for (int nt = 0; nt < 16; ++nt) {
        for (int i = t; i < 16 * 192; i += 128) {
            int n = i / 192, k = i % 192;
            w1T[n][k] = (_Float16)mw1[(size_t)k * 256 + nt * 16 + n];
        }
        __syncthreads();
        // preload all 6 B-fragments, then 6 back-to-back WMMAs
        v16h bfr[6];
#pragma unroll
        for (int k = 0; k < 6; ++k)
            bfr[k] = lds_b_frag(&w1T[0][0], 192, k * 32);
        float bv = smb1[nt * 16 + row];
        v8f acc = {bv, bv, bv, bv, bv, bv, bv, bv};
#pragma unroll
        for (int k = 0; k < 6; ++k)
            acc = wmma32(afrag[k], bfr[k], acc);
#pragma unroll
        for (int r = 0; r < 8; ++r)
            hF[wave * 16 + mbase + r][nt * 16 + row] = (_Float16)fmaxf(acc[r], 0.f);
        __syncthreads();
    }

    // tiny 256 -> 3 tail + residual add (VALU dots out of LDS)
    for (int o = t; o < 64 * 3; o += 128) {
        int n = o / 3, j = o % 3;
        float s = mb2[j];
        for (int c = 0; c < 256; ++c) s += (float)hF[n][c] * smw2[c * 3 + j];
        size_t g = (size_t)(node0 + n) * 3 + j;
        pwr[g] = points[g] + s;
    }
}

// ---------------- Plane fitting (all 32 planes independent; one block each) -
__device__ __forceinline__ float blockSum(float v, float* s) {
#pragma unroll
    for (int o = 16; o > 0; o >>= 1) v += __shfl_xor(v, o);
    int w = threadIdx.x >> 5, l = threadIdx.x & 31;
    __syncthreads();
    if (l == 0) s[w] = v;
    __syncthreads();
    float r = (threadIdx.x < 8) ? s[threadIdx.x] : 0.f;
    if (w == 0) {
#pragma unroll
        for (int o = 4; o > 0; o >>= 1) r += __shfl_xor(r, o);
        if (l == 0) s[0] = r;
    }
    __syncthreads();
    return s[0];
}

__global__ __launch_bounds__(256) void plane_fit_kernel(
    const float* __restrict__ pwr, const float* __restrict__ normals,
    const float* __restrict__ dists, float* __restrict__ planes)
{
    __shared__ float s[8];
    const int p = blockIdx.x;
    const float nx = normals[p * 3], ny = normals[p * 3 + 1], nz = normals[p * 3 + 2];
    const float d = dists[p];

    float c0 = 0.f, sx = 0.f, sy = 0.f, sz = 0.f;
    for (int i = threadIdx.x; i < NPTS; i += 256) {
        float x = pwr[i * 3], y = pwr[i * 3 + 1], z = pwr[i * 3 + 2];
        if (fabsf(x * nx + y * ny + z * nz - d) < 0.01f) {
            c0 += 1.f; sx += x; sy += y; sz += z;
        }
    }
    float cnt = blockSum(c0, s);
    sx = blockSum(sx, s); sy = blockSum(sy, s); sz = blockSum(sz, s);
    float inv = 1.f / fmaxf(cnt, 1.f);
    float cx = sx * inv, cy = sy * inv, cz = sz * inv;

    float m00 = 0, m01 = 0, m02 = 0, m11 = 0, m12 = 0, m22 = 0;
    for (int i = threadIdx.x; i < NPTS; i += 256) {
        float x = pwr[i * 3], y = pwr[i * 3 + 1], z = pwr[i * 3 + 2];
        if (fabsf(x * nx + y * ny + z * nz - d) < 0.01f) {
            float dx = x - cx, dy = y - cy, dz = z - cz;
            m00 += dx * dx; m01 += dx * dy; m02 += dx * dz;
            m11 += dy * dy; m12 += dy * dz; m22 += dz * dz;
        }
    }
    m00 = blockSum(m00, s); m01 = blockSum(m01, s); m02 = blockSum(m02, s);
    m11 = blockSum(m11, s); m12 = blockSum(m12, s); m22 = blockSum(m22, s);

    if (threadIdx.x == 0) {
        // smallest eigenvalue of symmetric 3x3 (trigonometric closed form)
        float ev;
        float p1 = m01 * m01 + m02 * m02 + m12 * m12;
        if (p1 < 1e-18f) {
            ev = fminf(m00, fminf(m11, m22));
        } else {
            float q = (m00 + m11 + m22) * (1.f / 3.f);
            float p2 = (m00 - q) * (m00 - q) + (m11 - q) * (m11 - q) +
                       (m22 - q) * (m22 - q) + 2.f * p1;
            float pp = sqrtf(p2 * (1.f / 6.f));
            float ip = 1.f / pp;
            float b00 = (m00 - q) * ip, b01 = m01 * ip, b02 = m02 * ip;
            float b11 = (m11 - q) * ip, b12 = m12 * ip, b22 = (m22 - q) * ip;
            float det = b00 * (b11 * b22 - b12 * b12) - b01 * (b01 * b22 - b12 * b02)
                      + b02 * (b01 * b12 - b11 * b02);
            float r = fminf(1.f, fmaxf(-1.f, det * 0.5f));
            float phi = acosf(r) * (1.f / 3.f);
            ev = q + 2.f * pp * cosf(phi + 2.09439510239f);  // smallest root
        }
        // eigenvector: best cross product of rows of (A - ev*I)
        float r0x = m00 - ev, r0y = m01,      r0z = m02;
        float r1x = m01,      r1y = m11 - ev, r1z = m12;
        float r2x = m02,      r2y = m12,      r2z = m22 - ev;
        float cAx = r0y * r1z - r0z * r1y, cAy = r0z * r1x - r0x * r1z, cAz = r0x * r1y - r0y * r1x;
        float cBx = r0y * r2z - r0z * r2y, cBy = r0z * r2x - r0x * r2z, cBz = r0x * r2y - r0y * r2x;
        float cCx = r1y * r2z - r1z * r2y, cCy = r1z * r2x - r1x * r2z, cCz = r1x * r2y - r1y * r2x;
        float nA = cAx * cAx + cAy * cAy + cAz * cAz;
        float nB = cBx * cBx + cBy * cBy + cBz * cBz;
        float nC = cCx * cCx + cCy * cCy + cCz * cCz;
        float vx = cAx, vy = cAy, vz = cAz, nn = nA;
        if (nB > nn) { vx = cBx; vy = cBy; vz = cBz; nn = nB; }
        if (nC > nn) { vx = cCx; vy = cCy; vz = cCz; nn = nC; }
        if (nn < 1e-30f) { vx = 1.f; vy = 0.f; vz = 0.f; nn = 1.f; }
        float rl = rsqrtf(nn);
        vx *= rl; vy *= rl; vz *= rl;
        if (vx * nx + vy * ny + vz * nz < 0.f) { vx = -vx; vy = -vy; vz = -vz; }
        planes[p * 8 + 0] = vx;
        planes[p * 8 + 1] = vy;
        planes[p * 8 + 2] = vz;
        planes[p * 8 + 3] = cx * vx + cy * vy + cz * vz;   // rd
        planes[p * 8 + 4] = (cnt >= 3.f) ? 1.f : 0.f;      // valid
    }
}

// ---------------- Sequential projection compose (per-point independent) -----
__global__ __launch_bounds__(256) void plane_apply_kernel(
    const float* __restrict__ pwr, const float* __restrict__ normals,
    const float* __restrict__ dists, const float* __restrict__ planes,
    float* __restrict__ out)
{
    int i = blockIdx.x * 256 + threadIdx.x;
    if (i >= NPTS) return;
    float px = pwr[i * 3], py = pwr[i * 3 + 1], pz = pwr[i * 3 + 2]; // mask source
    float qx = px, qy = py, qz = pz;                                  // projected
#pragma unroll 4
    for (int p = 0; p < 32; ++p) {
        float nx = normals[p * 3], ny = normals[p * 3 + 1], nz = normals[p * 3 + 2];
        bool mask = fabsf(px * nx + py * ny + pz * nz - dists[p]) < 0.01f;
        if (mask && planes[p * 8 + 4] > 0.f) {
            float rx = planes[p * 8], ry = planes[p * 8 + 1], rz = planes[p * 8 + 2];
            float s = qx * rx + qy * ry + qz * rz - planes[p * 8 + 3];
            qx -= rx * s; qy -= ry * s; qz -= rz * s;
        }
    }
    out[i * 3] = qx; out[i * 3 + 1] = qy; out[i * 3 + 2] = qz;
}

extern "C" void kernel_launch(void* const* d_in, const int* in_sizes, int n_in,
                              void* d_out, int out_size, void* d_ws, size_t ws_size,
                              hipStream_t stream) {
    const float* points  = (const float*)d_in[0];
    const float* normals = (const float*)d_in[1];
    const float* dists   = (const float*)d_in[2];
    const float* w1a = (const float*)d_in[3];  const float* b1a = (const float*)d_in[4];
    const float* w1b = (const float*)d_in[5];  const float* b1b = (const float*)d_in[6];
    const float* w2a = (const float*)d_in[7];  const float* b2a = (const float*)d_in[8];
    const float* w2b = (const float*)d_in[9];  const float* b2b = (const float*)d_in[10];
    const float* w3a = (const float*)d_in[11]; const float* b3a = (const float*)d_in[12];
    const float* w3b = (const float*)d_in[13]; const float* b3b = (const float*)d_in[14];
    const float* mw1 = (const float*)d_in[15]; const float* mb1 = (const float*)d_in[16];
    const float* mw2 = (const float*)d_in[17]; const float* mb2 = (const float*)d_in[18];
    const int*   ei  = (const int*)d_in[19];
    const int*   src = ei + (size_t)NPTS * KNN;   // row 1 of edge_index

    float* ws = (float*)d_ws;
    float* f1     = ws;
    float* f2     = f1 + (size_t)NPTS * 64;
    float* f3     = f2 + (size_t)NPTS * 64;
    float* pwr    = f3 + (size_t)NPTS * 64;
    float* planes = pwr + (size_t)NPTS * 3;

    edgeconv_kernel<3> <<<NPTS / 4, 128, 0, stream>>>(points, src, w1a, b1a, w1b, b1b, f1);
    edgeconv_kernel<64><<<NPTS / 4, 128, 0, stream>>>(f1,     src, w2a, b2a, w2b, b2b, f2);
    edgeconv_kernel<64><<<NPTS / 4, 128, 0, stream>>>(f2,     src, w3a, b3a, w3b, b3b, f3);
    mlp_kernel<<<NPTS / 64, 128, 0, stream>>>(f1, f2, f3, points, mw1, mb1, mw2, mb2, pwr);
    plane_fit_kernel<<<32, 256, 0, stream>>>(pwr, normals, dists, planes);
    plane_apply_kernel<<<NPTS / 256, 256, 0, stream>>>(pwr, normals, dists, planes, (float*)d_out);
}
